// SWA_16114717295080
// MI455X (gfx1250) — matrix operand order repaired and checked by reading, not compiled, not run on hardware
//
#include <hip/hip_runtime.h>
#include <hip/hip_bf16.h>

typedef __attribute__((ext_vector_type(16))) _Float16 v16h;
typedef __attribute__((ext_vector_type(8)))  _Float16 v8h;
typedef __attribute__((ext_vector_type(8)))  float    v8f;

#define WMMA_F16(A, B, C) \
  __builtin_amdgcn_wmma_f32_16x16x32_f16(false, (A), false, (B), (short)0, (C), false, false)

// Problem constants
#define HP   24          // H = W
#define HW   576         // H*W queries
#define CC   512         // channels
#define NHD  8           // heads
#define HD   64          // head dim
#define NPIX 900         // 30*30 padded pixels (distinct keys)
#define NPAD 928         // padded to 29 tiles of 32

// ---- WMMA fragment loaders (CDNA5 16-bit layouts, wave32) ----
// A 16x32 f16: lane l holds row M=l&15; K = kb..kb+7 and kb+16..kb+23, kb = (l&16)?8:0
__device__ inline v16h loadA(const _Float16* rowptr, int kchunk) {
  int lane = threadIdx.x & 31;
  int kb = kchunk + ((lane & 16) ? 8 : 0);
  v8h lo = *(const v8h*)(rowptr + kb);
  v8h hi = *(const v8h*)(rowptr + kb + 16);
  return __builtin_shufflevector(lo, hi, 0,1,2,3,4,5,6,7,8,9,10,11,12,13,14,15);
}
// B 32x16 f16: lane l holds col N=l&15; K = kb..kb+7 and kb+16..kb+23, kb = (l&16)?8:0
__device__ inline v16h loadB(const _Float16* colptr, int kchunk) {
  int lane = threadIdx.x & 31;
  int kb = kchunk + ((lane & 16) ? 8 : 0);
  v8h lo = *(const v8h*)(colptr + kb);
  v8h hi = *(const v8h*)(colptr + kb + 16);
  return __builtin_shufflevector(lo, hi, 0,1,2,3,4,5,6,7,8,9,10,11,12,13,14,15);
}

// Half-wave (16-lane) reductions: xor masks < 16 stay inside each half of the wave32
__device__ inline float hmax16(float v) {
  v = fmaxf(v, __shfl_xor(v, 1));
  v = fmaxf(v, __shfl_xor(v, 2));
  v = fmaxf(v, __shfl_xor(v, 4));
  v = fmaxf(v, __shfl_xor(v, 8));
  return v;
}
__device__ inline float hsum16(float v) {
  v += __shfl_xor(v, 1);
  v += __shfl_xor(v, 2);
  v += __shfl_xor(v, 4);
  v += __shfl_xor(v, 8);
  return v;
}

// multiplicity weight of padded key column n (duplication count in the 49-tap unfold)
__device__ inline float colw(int n) {
  if (n >= NPIX) return 0.f;
  int pi = n / 30, pj = n % 30;
  int wi = (pi < 6 ? pi : 6) - (pi - 23 > 0 ? pi - 23 : 0) + 1;
  int wj = (pj < 6 ? pj : 6) - (pj - 23 > 0 ? pj - 23 : 0) + 1;
  return (float)(wi * wj);
}

// ---------------- Kernel 1: reflect-pad + f16 conversion ----------------
__global__ void swa_prep(const float* __restrict__ x,
                         const float* __restrict__ Wq,
                         const float* __restrict__ Wk,
                         const float* __restrict__ Wv,
                         _Float16* __restrict__ xh,
                         _Float16* __restrict__ Whq,
                         _Float16* __restrict__ Whk,
                         _Float16* __restrict__ Whv) {
  int idx = blockIdx.x * 256 + threadIdx.x;
  const int n_xh = NPIX * CC;       // 460800
  const int n_w  = CC * CC;         // 262144
  if (idx < n_xh) {
    int p = idx >> 9;               // padded pixel 0..899
    int c = idx & 511;
    int si = p / 30 - 3, sj = p % 30 - 3;            // source coords in [-3,26]
    si = si < 0 ? -si : (si > 23 ? 46 - si : si);    // reflect (no edge repeat)
    sj = sj < 0 ? -sj : (sj > 23 ? 46 - sj : sj);
    xh[(size_t)p * CC + c] = (_Float16)x[(size_t)c * HW + si * HP + sj];
  } else {
    int t = idx - n_xh;
    if (t < n_w)            Whq[t]           = (_Float16)Wq[t];
    else if (t < 2 * n_w)   Whk[t - n_w]     = (_Float16)Wk[t - n_w];
    else if (t < 3 * n_w)   Whv[t - 2 * n_w] = (_Float16)Wv[t - 2 * n_w];
  }
}

// ---------------- Kernel 2: 1x1-conv GEMMs via WMMA ----------------
// out(512 x Npix) = W(512x512) * xh^T ; one wave = one 16x16 output tile.
__global__ __launch_bounds__(32) void swa_proj(
    const _Float16* __restrict__ xh,
    const _Float16* __restrict__ Whq, const _Float16* __restrict__ Whk,
    const _Float16* __restrict__ Whv,
    const float* __restrict__ bq, const float* __restrict__ bk,
    const float* __restrict__ bv,
    _Float16* __restrict__ Qb,   // [head][q(576)][d(64)]
    _Float16* __restrict__ Kb,   // [head][n(928)][d(64)]  (zero pad n>=900)
    _Float16* __restrict__ VT)   // [head][d(64)][n(928)]  (zero pad n>=900)
{
  int blk  = blockIdx.x;
  int lane = threadIdx.x & 31;
  int lcol = lane & 15;
  int half = (lane >> 4) & 1;

  int mode, otile, ptile;
  if (blk < 1152)      { mode = 0; otile = blk / 36; ptile = blk % 36; }          // Q: 32x36
  else if (blk < 3008) { int b = blk - 1152; mode = 1; otile = b / 58; ptile = b % 58; } // K
  else                 { int b = blk - 3008; mode = 2; otile = b / 58; ptile = b % 58; } // V

  const _Float16* W    = (mode == 0) ? Whq : ((mode == 1) ? Whk : Whv);
  const float*    bias = (mode == 0) ? bq  : ((mode == 1) ? bk  : bv);

  int orow = otile * 16 + lcol;            // A-fragment row (output channel)
  int p    = ptile * 16 + lcol;            // B-fragment column (pixel)
  int pp;                                  // pixel index into xh (padded space)
  if (mode == 0) { int qi = p / 24, qj = p % 24; pp = (qi + 3) * 30 + (qj + 3); }
  else           { pp = (p < NPIX) ? p : (NPIX - 1); }

  const _Float16* wrow = W  + (size_t)orow * CC;
  const _Float16* xcol = xh + (size_t)pp   * CC;

  v8f acc = {};
#pragma unroll 4
  for (int c = 0; c < CC; c += 32) {
    v16h a = loadA(wrow, c);
    v16h b = loadB(xcol, c);
    acc = WMMA_F16(a, b, acc);
  }

#pragma unroll
  for (int j = 0; j < 8; ++j) {
    int o = otile * 16 + j + 8 * half;     // C-frag row
    int h = o >> 6, d = o & 63;
    float val = acc[j] + bias[o];
    if (mode == 0) {
      int q = ptile * 16 + lcol;
      Qb[((size_t)(h * HW + q)) * HD + d] = (_Float16)val;
    } else {
      int n = ptile * 16 + lcol;
      _Float16 hv = (n < NPIX) ? (_Float16)val : (_Float16)0.f;
      if (mode == 1) Kb[((size_t)(h * NPAD + n)) * HD + d] = hv;
      else           VT[((size_t)(h * HD + d)) * NPAD + n] = hv;
    }
  }
}

// ---------------- Kernel 3: flash attention with multiplicity weights ----------------
// One wave per (head, 16-query tile). 29 key tiles of 32. 8 WMMAs per tile.
__global__ __launch_bounds__(32) void swa_attn(
    const _Float16* __restrict__ Qb, const _Float16* __restrict__ Kb,
    const _Float16* __restrict__ VT,
    const float* __restrict__ x, const float* __restrict__ gamma,
    float* __restrict__ out)
{
  __shared__ alignas(16) _Float16 Pl[16 * 32];   // P tile staging (C-frag -> A-frag)
  int head  = blockIdx.x / 36;
  int qtile = blockIdx.x % 36;
  int lane  = threadIdx.x & 31;
  int lcol  = lane & 15;
  int half  = (lane >> 4) & 1;

  const _Float16* Qrow = Qb + ((size_t)(head * HW + qtile * 16 + lcol)) * HD;
  v16h aq0 = loadA(Qrow, 0);
  v16h aq1 = loadA(Qrow, 32);

  v8f o0 = {}, o1 = {}, o2 = {}, o3 = {};
  float mrow[8], srow[8];
#pragma unroll
  for (int j = 0; j < 8; ++j) { mrow[j] = -1e30f; srow[j] = 0.f; }

  for (int kt = 0; kt < 29; ++kt) {
    int n0 = kt * 32;
    // E = Q * K^T for 16 queries x 32 keys (two 16-wide C frags)
    const _Float16* kc0 = Kb + ((size_t)(head * NPAD + n0 + lcol)) * HD;
    const _Float16* kc1 = kc0 + 16 * HD;
    v8f e0 = {}, e1 = {};
    e0 = WMMA_F16(aq0, loadB(kc0, 0),  e0);
    e0 = WMMA_F16(aq1, loadB(kc0, 32), e0);
    e1 = WMMA_F16(aq0, loadB(kc1, 0),  e1);
    e1 = WMMA_F16(aq1, loadB(kc1, 32), e1);

    float w0 = colw(n0 + lcol);        // duplication weight of this lane's two columns
    float w1 = colw(n0 + 16 + lcol);

    float alpha[8];
#pragma unroll
    for (int j = 0; j < 8; ++j) {
      float t = fmaxf(e0[j], e1[j]);
      t = hmax16(t);                                // row max over this 32-key tile
      float mnew = fmaxf(mrow[j], t);
      float a  = __expf(mrow[j] - mnew);
      float p0 = w0 * __expf(e0[j] - mnew);
      float p1 = w1 * __expf(e1[j] - mnew);
      srow[j]  = srow[j] * a + hsum16(p0 + p1);
      mrow[j]  = mnew;
      alpha[j] = a;
      int row = j + 8 * half;
      Pl[row * 32 + lcol]      = (_Float16)p0;
      Pl[row * 32 + 16 + lcol] = (_Float16)p1;
    }
    __syncthreads();

    // Re-read P in A-fragment layout, rescale O, accumulate P*V
    const _Float16* prow = &Pl[lcol * 32];
    v16h ap = loadA(prow, 0);
#pragma unroll
    for (int j = 0; j < 8; ++j) {
      o0[j] *= alpha[j]; o1[j] *= alpha[j]; o2[j] *= alpha[j]; o3[j] *= alpha[j];
    }
    const _Float16* vb = VT + (size_t)(head * HD) * NPAD + n0;
    o0 = WMMA_F16(ap, loadB(vb + (size_t)( 0 + lcol) * NPAD, 0), o0);
    o1 = WMMA_F16(ap, loadB(vb + (size_t)(16 + lcol) * NPAD, 0), o1);
    o2 = WMMA_F16(ap, loadB(vb + (size_t)(32 + lcol) * NPAD, 0), o2);
    o3 = WMMA_F16(ap, loadB(vb + (size_t)(48 + lcol) * NPAD, 0), o3);
    __syncthreads();
  }

  float g = gamma[0];
#pragma unroll
  for (int j = 0; j < 8; ++j) {
    int q = qtile * 16 + j + 8 * half;
    float inv = 1.f / srow[j];
    {
      size_t i0 = ((size_t)(head * HD +  0 + lcol)) * HW + q;
      size_t i1 = ((size_t)(head * HD + 16 + lcol)) * HW + q;
      size_t i2 = ((size_t)(head * HD + 32 + lcol)) * HW + q;
      size_t i3 = ((size_t)(head * HD + 48 + lcol)) * HW + q;
      out[i0] = g * (o0[j] * inv) + x[i0];
      out[i1] = g * (o1[j] * inv) + x[i1];
      out[i2] = g * (o2[j] * inv) + x[i2];
      out[i3] = g * (o3[j] * inv) + x[i3];
    }
  }
}

extern "C" void kernel_launch(void* const* d_in, const int* in_sizes, int n_in,
                              void* d_out, int out_size, void* d_ws, size_t ws_size,
                              hipStream_t stream) {
  const float* x     = (const float*)d_in[0];
  const float* Wq    = (const float*)d_in[1];
  const float* bq    = (const float*)d_in[2];
  const float* Wk    = (const float*)d_in[3];
  const float* bk    = (const float*)d_in[4];
  const float* Wv    = (const float*)d_in[5];
  const float* bv    = (const float*)d_in[6];
  const float* gamma = (const float*)d_in[7];
  float* out = (float*)d_out;

  char* ws = (char*)d_ws;
  // workspace layout (bytes)
  _Float16* xh  = (_Float16*)(ws);                       // 900*512*2  = 921600
  _Float16* Whq = (_Float16*)(ws + 921600);              // 512*512*2  = 524288
  _Float16* Whk = (_Float16*)(ws + 921600 + 524288);
  _Float16* Whv = (_Float16*)(ws + 921600 + 1048576);
  _Float16* Qb  = (_Float16*)(ws + 2494464);             // 8*576*64*2 = 589824
  _Float16* Kb  = (_Float16*)(ws + 3084288);             // 8*928*64*2 = 950272
  _Float16* VT  = (_Float16*)(ws + 4034560);             // 8*64*928*2 = 950272
  (void)in_sizes; (void)n_in; (void)out_size; (void)ws_size;

  const int totalPrep = NPIX * CC + 3 * CC * CC;         // 1,247,232
  swa_prep<<<(totalPrep + 255) / 256, 256, 0, stream>>>(x, Wq, Wk, Wv, xh, Whq, Whk, Whv);
  swa_proj<<<1152 + 1856 + 1856, 32, 0, stream>>>(xh, Whq, Whk, Whv, bq, bk, bv, Qb, Kb, VT);
  swa_attn<<<NHD * 36, 32, 0, stream>>>(Qb, Kb, VT, x, gamma, out);
}
